// RWKV6Layer_2113123909865
// MI455X (gfx1250) — matrix operand order repaired and checked by reading, not compiled, not run on hardware
//
#include <hip/hip_runtime.h>

// ---------------------------------------------------------------------------
// RWKV6 layer for MI455X (gfx1250, wave32).
// GEMMs: V_WMMA_F32_16X16X32_BF16, operands pre-swizzled to ISA fragment
// layouts; inner loop runs a 2-deep software pipeline with statically named
// fragment registers (no runtime-indexed arrays -> no scratch spills).
// Each wave owns a 64x16 output strip (4 m-tiles share each B fragment).
// WKV recurrence: 3-pass chunked affine scan (16 chunks of 256 steps) so the
// sequential part runs on 512 workgroups instead of 32.
// ---------------------------------------------------------------------------

#define DEV __device__ __forceinline__

constexpr int B  = 8;
constexpr int T  = 4096;
constexpr int C  = 512;
constexpr int KD = 256;
constexpr int VD = 512;
constexpr int H  = 4;
constexpr int HK = 64;    // head qk dim
constexpr int HV = 128;   // head v dim
constexpr int LR = 32;    // proj low rank
constexpr int GLR= 64;    // gate low rank
constexpr int FF = 1024;
constexpr int M  = B * T; // 32768 token rows
constexpr int CT = 256;   // WKV scan chunk length
constexpr int NC = T / CT;// 16 chunks

// ---- bf16 <-> f32 helpers (storage = ushort; RNE rounding) ----------------
DEV unsigned short f2bf(float f) {
  unsigned u = __builtin_bit_cast(unsigned, f);
  return (unsigned short)((u + 0x7fffu + ((u >> 16) & 1u)) >> 16);
}
DEV float bf2f(unsigned short h) {
  unsigned u = ((unsigned)h) << 16;
  return __builtin_bit_cast(float, u);
}
DEV float fast_tanh(float x) {
  x = fminf(fmaxf(x, -15.f), 15.f);
  float e = __expf(2.f * x);
  return (e - 1.f) / (e + 1.f);
}

// ---- WMMA types ------------------------------------------------------------
typedef __bf16 v16bf __attribute__((ext_vector_type(16)));
typedef float  v8f   __attribute__((ext_vector_type(8)));
struct alignas(16) Frag { unsigned int r[8]; };   // 32B = one 16-bf16 fragment

// ---- fragment-layout addressing (ISA 7.12.2, 16-bit A / B matrices) --------
DEV size_t afrag_off(int m, int n, int nChunks) {
  int kk   = n & 31;
  int lane = (m & 15) + ((kk & 8) ? 16 : 0);
  int elem = (kk & 7) + ((kk & 16) ? 8 : 0);
  return ((((size_t)(m >> 4) * nChunks + (n >> 5)) * 32) + lane) * 16 + elem;
}
DEV void st_afrag(unsigned short* buf, int nChunks, int m, int n, unsigned short v) {
  buf[afrag_off(m, n, nChunks)] = v;
}

// ===========================================================================
// Weight convert f32 -> bf16 directly into B-fragment layout.
// ===========================================================================
__global__ __launch_bounds__(256) void k_cvtB(const float* __restrict__ s,
                                              unsigned short* __restrict__ d,
                                              int K, int N) {
  int i = blockIdx.x * 256 + threadIdx.x;
  if (i >= K * N) return;
  int k = i / N, n = i - k * N;
  int kk   = k & 31;
  int lane = (n & 15) + ((kk & 8) ? 16 : 0);
  int elem = (kk & 7) + ((kk & 16) ? 8 : 0);
  size_t off = ((((size_t)(n >> 4) * (K >> 5) + (k >> 5)) * 32) + lane) * 16 + elem;
  d[off] = f2bf(s[i]);
}

// ===========================================================================
// LayerNorm: f32 [M,C] -> bf16 [M,C] (row-major: MIX epilogue needs shift)
// ===========================================================================
__global__ __launch_bounds__(256) void k_ln(const float* __restrict__ x,
                                            const float* __restrict__ w,
                                            const float* __restrict__ b,
                                            unsigned short* __restrict__ out) {
  int m = blockIdx.x, tid = threadIdx.x;
  const float* row = x + (size_t)m * C;
  float s0 = 0.f, s1 = 0.f;
  for (int c = tid; c < C; c += 256) { float v = row[c]; s0 += v; s1 += v * v; }
  __shared__ float r0[256], r1[256];
  r0[tid] = s0; r1[tid] = s1; __syncthreads();
  for (int s = 128; s > 0; s >>= 1) {
    if (tid < s) { r0[tid] += r0[tid + s]; r1[tid] += r1[tid + s]; }
    __syncthreads();
  }
  float mean = r0[0] * (1.f / C);
  float var  = r1[0] * (1.f / C) - mean * mean;
  float inv  = rsqrtf(var + 1e-5f);
  for (int c = tid; c < C; c += 256)
    out[(size_t)m * C + c] = f2bf((row[c] - mean) * inv * w[c] + b[c]);
}

// ===========================================================================
// Constant-mu token-shift lerp: z = h + (shift(h)-h)*mu -> A-fragment bf16
// ===========================================================================
__global__ __launch_bounds__(256) void k_mixc(const unsigned short* __restrict__ h,
                                              const float* __restrict__ mu,
                                              unsigned short* __restrict__ z) {
  int m = blockIdx.x, tid = threadIdx.x;
  bool first = (m & (T - 1)) == 0;
  for (int c = tid; c < C; c += 256) {
    float hc = bf2f(h[(size_t)m * C + c]);
    float hp = first ? 0.f : bf2f(h[(size_t)(m - 1) * C + c]);
    st_afrag(z, C / 32, m, c, f2bf(hc + (hp - hc) * mu[c]));
  }
}

// ===========================================================================
// WMMA GEMM on pre-swizzled fragments, 2-deep pipelined K loop.
//   block = 128 = 4 waves; each wave computes a 64x16 strip.
//   grid = (M/256, N/16).  EXEC always all-ones at the WMMAs.
// ===========================================================================
enum { GM_F32 = 0, GM_TANH_AF = 2, GM_MIX_AF = 3, GM_NEGEXP = 4,
       GM_RELU2_AF = 5, GM_SIG = 6, GM_RES = 7, GM_RESGATE = 8 };

#define LOAD_CHUNK(kc_, fb_, fa_)                                      \
  {                                                                    \
    const uint4* bp_ = bbase + (size_t)(kc_) * 64;                     \
    *(uint4*)&fb_.r[0] = bp_[0];                                       \
    *(uint4*)&fb_.r[4] = bp_[1];                                       \
    __builtin_prefetch(bp_ + 128, 0, 1);                               \
    _Pragma("unroll")                                                  \
    for (int i_ = 0; i_ < 4; ++i_) {                                   \
      const uint4* ap_ = abase[i_] + (size_t)(kc_) * 64;               \
      *(uint4*)&fa_[i_].r[0] = ap_[0];                                 \
      *(uint4*)&fa_[i_].r[4] = ap_[1];                                 \
    }                                                                  \
  }

#define WMMA4(fa_, fb_)                                                \
  _Pragma("unroll")                                                    \
  for (int i_ = 0; i_ < 4; ++i_)                                       \
    acc[i_] = __builtin_amdgcn_wmma_f32_16x16x32_bf16(                 \
        false, __builtin_bit_cast(v16bf, fa_[i_]),                     \
        false, __builtin_bit_cast(v16bf, fb_),                         \
        (short)0, acc[i_], false, false);

template <int MODE>
__global__ __launch_bounds__(128)
void k_gemm(const unsigned short* __restrict__ Af, int aChunks, int aBase,
            const unsigned short* __restrict__ Bf,
            const float* __restrict__ bias,        // [N]   (MIX, NEGEXP)
            const float* __restrict__ resid,       // [M,N] (RES, RESGATE)
            const unsigned short* __restrict__ hln,// [M,N] row-major (MIX)
            const float* __restrict__ gate,        // [M,N] (RESGATE)
            float* __restrict__ outF,
            unsigned short* __restrict__ outB, int outChunks,
            int K, int N) {
  const int wave = threadIdx.x >> 5;
  const int lane = threadIdx.x & 31;
  const int mt0  = blockIdx.x * 16 + wave * 4;   // first of 4 m-tiles
  const int nt   = blockIdx.y;
  const int KC   = K >> 5;
  __builtin_assume(KC >= 1);

  v8f acc[4];
#pragma unroll
  for (int i = 0; i < 4; ++i) acc[i] = (v8f){0.f,0.f,0.f,0.f,0.f,0.f,0.f,0.f};

  const uint4* bbase = (const uint4*)Bf + ((size_t)nt * KC * 32 + lane) * 2;
  const uint4* abase[4];
#pragma unroll
  for (int i = 0; i < 4; ++i)
    abase[i] = (const uint4*)Af +
        ((((size_t)(mt0 + i) * aChunks + aBase) * 32) + lane) * 2;

  // Two statically named pipeline stages: X and Y (constant indices only,
  // so everything stays in VGPRs).
  Frag fbX, fbY, faX[4], faY[4];
  LOAD_CHUNK(0, fbX, faX);
  int kc = 0;
  for (; kc + 2 <= KC; kc += 2) {
    LOAD_CHUNK(kc + 1, fbY, faY);
    WMMA4(faX, fbX);
    if (kc + 2 < KC) LOAD_CHUNK(kc + 2, fbX, faX);
    WMMA4(faY, fbY);
  }
  if (kc < KC) { WMMA4(faX, fbX); }    // odd tail (KC == 1 case)

  const int ncol = nt * 16 + (lane & 15);
  const int hi8  = (lane < 16) ? 0 : 8;
#pragma unroll
  for (int i = 0; i < 4; ++i) {
#pragma unroll
    for (int r = 0; r < 8; ++r) {
      const int mm = (mt0 + i) * 16 + r + hi8;
      const size_t idx = (size_t)mm * N + ncol;
      float a = acc[i][r];
      if constexpr (MODE == GM_F32) {
        outF[idx] = a;
      } else if constexpr (MODE == GM_TANH_AF) {
        st_afrag(outB, outChunks, mm, ncol, f2bf(fast_tanh(a)));
      } else if constexpr (MODE == GM_MIX_AF) {
        float mus = a + bias[ncol];
        float hc  = bf2f(hln[idx]);
        float hp  = ((mm & (T - 1)) == 0) ? 0.f : bf2f(hln[idx - N]);
        st_afrag(outB, outChunks, mm, ncol, f2bf(hc + (hp - hc) * mus));
      } else if constexpr (MODE == GM_NEGEXP) {
        outF[idx] = -__expf(a + bias[ncol]);
      } else if constexpr (MODE == GM_RELU2_AF) {
        float t = fmaxf(a, 0.f);
        st_afrag(outB, outChunks, mm, ncol, f2bf(t * t));
      } else if constexpr (MODE == GM_SIG) {
        outF[idx] = 1.f / (1.f + __expf(-a));
      } else if constexpr (MODE == GM_RES) {
        outF[idx] = resid[idx] + a;
      } else if constexpr (MODE == GM_RESGATE) {
        outF[idx] = resid[idx] + gate[idx] * a;
      }
    }
  }
}

// ===========================================================================
// WKV6 chunked affine scan:  S_t = diag(e^{w_t}) S_{t-1} + k_t v_t^T
// Pass 1: per chunk, local accumulation A_c and per-row decay product D_c.
//         grid=(B,H,NC), block=128 (one lane per v-column).
// ===========================================================================
__global__ __launch_bounds__(128)
void k_wkv_chunk(const float* __restrict__ kk, const float* __restrict__ v,
                 const float* __restrict__ wd,
                 float* __restrict__ Abuf, float* __restrict__ Dbuf) {
  const int b = blockIdx.x, hh = blockIdx.y, c = blockIdx.z, vi = threadIdx.x;
  __shared__ float sk[HK], sw[HK];
  float S[HK];
#pragma unroll
  for (int i = 0; i < HK; ++i) S[i] = 0.f;
  float dp = 1.f;

  for (int t = c * CT; t < (c + 1) * CT; ++t) {
    const size_t row = (size_t)b * T + t;
    if (vi < HK) {
      const size_t kidx = row * KD + hh * HK + vi;
      sk[vi] = kk[kidx];
      float ew = __expf(wd[kidx]);
      sw[vi] = ew;
      dp *= ew;
    }
    __syncthreads();
    const float vt = v[row * VD + hh * HV + vi];
#pragma unroll
    for (int i = 0; i < HK; ++i) S[i] = sw[i] * S[i] + sk[i] * vt;
    __syncthreads();
  }
  const size_t base = (((size_t)(b * H + hh) * NC + c) * HK) * HV;
#pragma unroll
  for (int i = 0; i < HK; ++i) Abuf[base + (size_t)i * HV + vi] = S[i];
  if (vi < HK) Dbuf[((size_t)(b * H + hh) * NC + c) * HK + vi] = dp;
}

// Pass 2: sequential chunk combine; stores every chunk's start state.
//         grid=(B*H), block=128.
__global__ __launch_bounds__(128)
void k_wkv_scan(const float* __restrict__ Abuf, const float* __restrict__ Dbuf,
                float* __restrict__ Sbuf) {
  const int bh = blockIdx.x, vi = threadIdx.x;
  __shared__ float sd[HK];
  float S[HK];
#pragma unroll
  for (int i = 0; i < HK; ++i) S[i] = 0.f;
  for (int c = 0; c < NC; ++c) {
    const size_t base = (((size_t)bh * NC + c) * HK) * HV;
#pragma unroll
    for (int i = 0; i < HK; ++i) Sbuf[base + (size_t)i * HV + vi] = S[i];
    if (vi < HK) sd[vi] = Dbuf[((size_t)bh * NC + c) * HK + vi];
    __syncthreads();
#pragma unroll
    for (int i = 0; i < HK; ++i)
      S[i] = sd[i] * S[i] + Abuf[base + (size_t)i * HV + vi];
    __syncthreads();
  }
}

// Pass 3: replay each chunk from its start state, producing outputs.
//         grid=(B,H,NC), block=128.
__global__ __launch_bounds__(128)
void k_wkv_out(const float* __restrict__ r, const float* __restrict__ kk,
               const float* __restrict__ v, const float* __restrict__ wd,
               const float* __restrict__ u, const float* __restrict__ Sbuf,
               float* __restrict__ o) {
  const int b = blockIdx.x, hh = blockIdx.y, c = blockIdx.z, vi = threadIdx.x;
  __shared__ float sr[HK], sk[HK], sw[HK], su[HK];
  float S[HK];
  const size_t base = (((size_t)(b * H + hh) * NC + c) * HK) * HV;
#pragma unroll
  for (int i = 0; i < HK; ++i) S[i] = Sbuf[base + (size_t)i * HV + vi];
  if (vi < HK) su[vi] = u[hh * HK + vi];
  __syncthreads();

  for (int t = c * CT; t < (c + 1) * CT; ++t) {
    const size_t row = (size_t)b * T + t;
    if (vi < HK) {
      const size_t kidx = row * KD + hh * HK + vi;
      sr[vi] = r[kidx];
      sk[vi] = kk[kidx];
      sw[vi] = __expf(wd[kidx]);
    }
    __syncthreads();
    const float vt = v[row * VD + hh * HV + vi];
    float out = 0.f;
#pragma unroll
    for (int i = 0; i < HK; ++i) {
      float kv = sk[i] * vt;
      out += sr[i] * (S[i] + su[i] * kv);
      S[i] = sw[i] * S[i] + kv;
    }
    o[row * VD + hh * HV + vi] = out;
    __syncthreads();
  }
}

// ===========================================================================
// Per-head GroupNorm * SiLU(gate) -> bf16 A-fragments. grid=(M,H), block=128.
// ===========================================================================
__global__ __launch_bounds__(128)
void k_gn_gate(const float* __restrict__ o, const float* __restrict__ g,
               const float* __restrict__ gw, const float* __restrict__ gb,
               unsigned short* __restrict__ obar) {
  const int m = blockIdx.x, hh = blockIdx.y, c = threadIdx.x;
  const int ch = hh * HV + c;
  const size_t idx = (size_t)m * VD + ch;
  float x = o[idx];
  __shared__ float red[128];
  red[c] = x; __syncthreads();
  for (int s = 64; s > 0; s >>= 1) { if (c < s) red[c] += red[c + s]; __syncthreads(); }
  float mu = red[0] * (1.f / HV); __syncthreads();
  float d = x - mu; red[c] = d * d; __syncthreads();
  for (int s = 64; s > 0; s >>= 1) { if (c < s) red[c] += red[c + s]; __syncthreads(); }
  float xn = d * rsqrtf(red[0] * (1.f / HV) + 1e-5f);
  float gv = g[idx];
  float silu = gv / (1.f + __expf(-gv));
  st_afrag(obar, VD / 32, m, ch, f2bf((xn * gw[ch] + gb[ch]) * silu));
}

// ===========================================================================
// Workspace arena (static layout, later stages reuse dead regions)
// ===========================================================================
#define SZ(n, b) ((size_t)(n) * (b))
constexpr size_t o_wx1  = 0;
constexpr size_t o_wx2  = o_wx1  + SZ(C * 160, 2);
constexpr size_t o_wr   = o_wx2  + SZ(5 * LR * C, 2);
constexpr size_t o_wk   = o_wr   + SZ(C * KD, 2);
constexpr size_t o_wv   = o_wk   + SZ(C * KD, 2);
constexpr size_t o_wg   = o_wv   + SZ(C * VD, 2);
constexpr size_t o_ww1  = o_wg   + SZ(C * VD, 2);
constexpr size_t o_ww2  = o_ww1  + SZ(C * GLR, 2);
constexpr size_t o_wo   = o_ww2  + SZ(GLR * KD, 2);
constexpr size_t o_wkey = o_wo   + SZ(VD * C, 2);
constexpr size_t o_wrec = o_wkey + SZ(C * FF, 2);
constexpr size_t o_wval = o_wrec + SZ(C * C, 2);
constexpr size_t o_h    = o_wval + SZ(FF * C, 2);     // h bf16; later h2 f32 spans h+zmu
constexpr size_t o_zmu  = o_h    + SZ(M * C, 2);
constexpr size_t o_xx   = o_zmu  + SZ(M * C, 2);
constexpr size_t o_z    = o_xx   + SZ(M * 160, 2);    // z0..z4 arena (167.8MB)
constexpr size_t o_r    = o_z    + 5 * SZ(M * C, 2);  // later hn
constexpr size_t o_k    = o_r    + SZ(M * KD, 4);     // later zk
constexpr size_t o_v    = o_k    + SZ(M * KD, 4);     // later kf
constexpr size_t o_g    = o_v    + SZ(M * VD, 4);     // later rf
constexpr size_t o_wl   = o_g    + SZ(M * VD, 4);
constexpr size_t o_wd   = o_wl   + SZ(M * GLR, 2);    // later zr
constexpr size_t WS_NEED= o_wd   + SZ(M * KD, 4);     // ~467 MB
// WKV-stage overlay inside the (dead) z arena:
constexpr size_t o_o    = o_z;                                // f32 [M,VD]
constexpr size_t o_obar = o_o    + SZ(M * VD, 4);             // bf16 A-frag
constexpr size_t o_wkA  = o_obar + SZ(M * VD, 2);             // f32 [B*H,NC,HK,HV]
constexpr size_t o_wkD  = o_wkA  + SZ(B * H * NC * HK * HV, 4);
constexpr size_t o_wkS  = o_wkD  + SZ(B * H * NC * HK, 4);
static_assert(o_wkS + SZ(B * H * NC * HK * HV, 4) <= o_r, "z arena overflow");

extern "C" void kernel_launch(void* const* d_in, const int* in_sizes, int n_in,
                              void* d_out, int out_size, void* d_ws, size_t ws_size,
                              hipStream_t stream) {
  const float* x      = (const float*)d_in[0];
  const float* ln1w   = (const float*)d_in[1];
  const float* ln1b   = (const float*)d_in[2];
  const float* mu_x   = (const float*)d_in[3];
  const float* W_x1   = (const float*)d_in[4];
  const float* W_x2   = (const float*)d_in[5];
  const float* x_bias = (const float*)d_in[6];
  const float* W_r    = (const float*)d_in[7];
  const float* W_k    = (const float*)d_in[8];
  const float* W_v    = (const float*)d_in[9];
  const float* W_g    = (const float*)d_in[10];
  const float* W_w1   = (const float*)d_in[11];
  const float* W_w2   = (const float*)d_in[12];
  const float* b_w2   = (const float*)d_in[13];
  const float* u      = (const float*)d_in[14];
  const float* gnw    = (const float*)d_in[15];
  const float* gnb    = (const float*)d_in[16];
  const float* W_o    = (const float*)d_in[17];
  const float* ln2w   = (const float*)d_in[18];
  const float* ln2b   = (const float*)d_in[19];
  const float* mu_k   = (const float*)d_in[20];
  const float* W_key  = (const float*)d_in[21];
  const float* mu_r   = (const float*)d_in[22];
  const float* W_rec  = (const float*)d_in[23];
  const float* W_val  = (const float*)d_in[24];
  char* ws = (char*)d_ws;
  float* out = (float*)d_out;

  auto BF = [&](size_t off) { return (unsigned short*)(ws + off); };
  auto FP = [&](size_t off) { return (float*)(ws + off); };

  // --- 1. weights -> bf16 in B-fragment layout ------------------------------
  auto cvtB = [&](const float* s, size_t off, int K, int N) {
    size_t n = (size_t)K * N;
    k_cvtB<<<dim3((unsigned)((n + 255) / 256)), 256, 0, stream>>>(s, BF(off), K, N);
  };
  cvtB(W_x1, o_wx1, C, 160);
  for (int i = 0; i < 5; ++i)
    cvtB(W_x2 + (size_t)i * LR * C, o_wx2 + (size_t)i * SZ(LR * C, 2), LR, C);
  cvtB(W_r,  o_wr,  C, KD);     cvtB(W_k,  o_wk,  C, KD);
  cvtB(W_v,  o_wv,  C, VD);     cvtB(W_g,  o_wg,  C, VD);
  cvtB(W_w1, o_ww1, C, GLR);    cvtB(W_w2, o_ww2, GLR, KD);
  cvtB(W_o,  o_wo,  VD, C);     cvtB(W_key,o_wkey, C, FF);
  cvtB(W_rec,o_wrec, C, C);     cvtB(W_val,o_wval, FF, C);

  auto grid = [&](int N) { return dim3((unsigned)(M / 256), (unsigned)(N / 16)); };

  // --- 2. attention LN + LoRA lerp factors ---------------------------------
  k_ln<<<M, 256, 0, stream>>>(x, ln1w, ln1b, BF(o_h));
  k_mixc<<<M, 256, 0, stream>>>(BF(o_h), mu_x, BF(o_zmu));
  k_gemm<GM_TANH_AF><<<grid(160), 128, 0, stream>>>(      // xx = tanh(zmu@W_x1)
      BF(o_zmu), C/32, 0, BF(o_wx1), nullptr, nullptr, nullptr, nullptr,
      nullptr, BF(o_xx), 160/32, C, 160);
  for (int i = 0; i < 5; ++i) {                           // z_i (MIX epilogue)
    k_gemm<GM_MIX_AF><<<grid(C), 128, 0, stream>>>(
        BF(o_xx), 160/32, i, BF(o_wx2) + (size_t)i * LR * C,
        x_bias + (size_t)i * C, nullptr, BF(o_h), nullptr,
        nullptr, BF(o_z) + (size_t)i * M * C, C/32, LR, C);
  }

  // --- 3. projections -------------------------------------------------------
  auto Z = [&](int i) { return BF(o_z) + (size_t)i * M * C; };
  k_gemm<GM_F32><<<grid(KD), 128, 0, stream>>>(           // r
      Z(0), C/32, 0, BF(o_wr), nullptr, nullptr, nullptr, nullptr,
      FP(o_r), nullptr, 0, C, KD);
  k_gemm<GM_TANH_AF><<<grid(GLR), 128, 0, stream>>>(      // decay LoRA stage 1
      Z(1), C/32, 0, BF(o_ww1), nullptr, nullptr, nullptr, nullptr,
      nullptr, BF(o_wl), GLR/32, C, GLR);
  k_gemm<GM_NEGEXP><<<grid(KD), 128, 0, stream>>>(        // w = -exp(lora@W_w2+b)
      BF(o_wl), GLR/32, 0, BF(o_ww2), b_w2, nullptr, nullptr, nullptr,
      FP(o_wd), nullptr, 0, GLR, KD);
  k_gemm<GM_F32><<<grid(KD), 128, 0, stream>>>(           // k
      Z(2), C/32, 0, BF(o_wk), nullptr, nullptr, nullptr, nullptr,
      FP(o_k), nullptr, 0, C, KD);
  k_gemm<GM_F32><<<grid(VD), 128, 0, stream>>>(           // v
      Z(3), C/32, 0, BF(o_wv), nullptr, nullptr, nullptr, nullptr,
      FP(o_v), nullptr, 0, C, VD);
  k_gemm<GM_F32><<<grid(VD), 128, 0, stream>>>(           // g
      Z(4), C/32, 0, BF(o_wg), nullptr, nullptr, nullptr, nullptr,
      FP(o_g), nullptr, 0, C, VD);

  // --- 4. WKV chunked scan (z arena dead -> o/obar/A/D/S overlays) ----------
  k_wkv_chunk<<<dim3(B, H, NC), 128, 0, stream>>>(
      FP(o_k), FP(o_v), FP(o_wd), FP(o_wkA), FP(o_wkD));
  k_wkv_scan<<<dim3(B * H), 128, 0, stream>>>(FP(o_wkA), FP(o_wkD), FP(o_wkS));
  k_wkv_out<<<dim3(B, H, NC), 128, 0, stream>>>(
      FP(o_r), FP(o_k), FP(o_v), FP(o_wd), u, FP(o_wkS), FP(o_o));
  k_gn_gate<<<dim3(M, H), 128, 0, stream>>>(FP(o_o), FP(o_g), gnw, gnb, BF(o_obar));

  // --- 5. output projection + residual (h2 reuses h+zmu span) ---------------
  const size_t o_h2 = o_h;                                // f32 [M,C]
  k_gemm<GM_RES><<<grid(C), 128, 0, stream>>>(
      BF(o_obar), VD/32, 0, BF(o_wo), nullptr, x, nullptr, nullptr,
      FP(o_h2), nullptr, 0, VD, C);

  // --- 6. channel mixing FFN ------------------------------------------------
  const size_t o_hn = o_r;   // bf16 [M,C] row-major
  const size_t o_zk = o_k;   // bf16 A-frag [M,C]
  const size_t o_zr = o_wd;  // bf16 A-frag [M,C]
  const size_t o_kf = o_v;   // bf16 A-frag [M,FF]
  const size_t o_rf = o_g;   // f32  [M,C]
  k_ln<<<M, 256, 0, stream>>>(FP(o_h2), ln2w, ln2b, BF(o_hn));
  k_mixc<<<M, 256, 0, stream>>>(BF(o_hn), mu_k, BF(o_zk));
  k_mixc<<<M, 256, 0, stream>>>(BF(o_hn), mu_r, BF(o_zr));
  k_gemm<GM_RELU2_AF><<<grid(FF), 128, 0, stream>>>(      // kf = relu(zk@W_key)^2
      BF(o_zk), C/32, 0, BF(o_wkey), nullptr, nullptr, nullptr, nullptr,
      nullptr, BF(o_kf), FF/32, C, FF);
  k_gemm<GM_SIG><<<grid(C), 128, 0, stream>>>(            // rf = sigmoid(zr@W_rec)
      BF(o_zr), C/32, 0, BF(o_wrec), nullptr, nullptr, nullptr, nullptr,
      FP(o_rf), nullptr, 0, C, C);
  k_gemm<GM_RESGATE><<<grid(C), 128, 0, stream>>>(        // out = h2 + rf*(kf@W_val)
      BF(o_kf), FF/32, 0, BF(o_wval), nullptr, FP(o_h2), nullptr, FP(o_rf),
      out, nullptr, 0, FF, C);

  (void)in_sizes; (void)n_in; (void)out_size; (void)ws_size; (void)WS_NEED;
}
